// VanillaRNN_57458072486002
// MI455X (gfx1250) — compile-verified, matrix operand order checked by reading, not run on hardware
//
#include <hip/hip_runtime.h>
#include <hip/hip_bf16.h>
#include <math.h>

// Problem constants (match reference).
#define B_  256
#define T_  512
#define D_  128
#define H_  2048
#define C_  128

typedef __bf16 bf16_t;
typedef __attribute__((ext_vector_type(16))) __bf16 v16bf;
typedef __attribute__((ext_vector_type(8)))  __bf16 v8bf;
typedef __attribute__((ext_vector_type(8)))  float  v8f;

// Branch-free fast tanh: output truncated to bf16 anyway, so ~1-ulp-f32
// accuracy is plenty. t = exp2(-2*log2e*|x|) in (0,1] -> no overflow/NaN.
//   tanh(x) = sign(x) * (1-t)/(1+t)
// Lowers to: v_exp_f32 + v_rcp_f32 + a few VALU, no exec-mask branches.
__device__ __forceinline__ float tanh_fast(float x) {
    const float ax = __builtin_fabsf(x);
    const float t  = __builtin_amdgcn_exp2f(ax * -2.885390081777927f); // -2*log2(e)
    const float r  = __builtin_amdgcn_rcpf(1.0f + t);
    return __builtin_copysignf((1.0f - t) * r, x);
}

// ---------------------------------------------------------------------------
// Pack W_hh [H,H] f32 row-major -> bf16 B-fragments for v_wmma_f32_16x16x32_bf16.
// Fragment (kc, nt) covers K rows [kc*32, kc*32+32), N cols [nt*16, nt*16+16).
// Per-lane layout (mirrors ISA 16-bit operand layout):
//   lane l, 16 contiguous bf16 at frag*512 + l*16:
//     j in [0,8):  K = kc*32 + (l>=16 ? 8:0) + j
//     j in [8,16): K = kc*32 + (l>=16 ? 8:0) + 16 + (j-8)   (== j+8)
//     N = nt*16 + (l & 15)
// ---------------------------------------------------------------------------
__global__ void pack_whh_kernel(const float* __restrict__ W, bf16_t* __restrict__ out) {
    int e = blockIdx.x * 256 + threadIdx.x;          // e < H_*H_
    int frag = e >> 9;                               // 512 elems / fragment
    int pos  = e & 511;
    int lane = pos >> 4;
    int j    = pos & 15;
    int kc   = frag >> 7;                            // / (H_/16 == 128)
    int nt   = frag & 127;
    int k = kc * 32 + ((lane >> 4) << 3) + ((j < 8) ? j : (j + 8));
    int n = (nt << 4) + (lane & 15);
    out[e] = (bf16_t)W[k * H_ + n];
}

// Same packing for W_ph [H, C] (C/16 == 8 N-tiles).
__global__ void pack_wph_kernel(const float* __restrict__ W, bf16_t* __restrict__ out) {
    int e = blockIdx.x * 256 + threadIdx.x;          // e < H_*C_
    int frag = e >> 9;
    int pos  = e & 511;
    int lane = pos >> 4;
    int j    = pos & 15;
    int kc   = frag >> 3;                            // / (C_/16 == 8)
    int nt   = frag & 7;
    int k = kc * 32 + ((lane >> 4) << 3) + ((j < 8) ? j : (j + 8));
    int n = (nt << 4) + (lane & 15);
    out[e] = (bf16_t)W[k * C_ + n];
}

// Zero h0 (bf16 zeros == 0x0000): count dwords.
__global__ void zero_kernel(unsigned int* __restrict__ p, int n_dwords) {
    int i = blockIdx.x * 256 + threadIdx.x;
    if (i < n_dwords) p[i] = 0u;
}

// Load a 16x32 bf16 A-fragment row chunk for this lane:
// two contiguous 8-element (16B) loads at +0 and +16 elements
// (compiler merges adjacent unrolled iterations into b128 clauses).
__device__ __forceinline__ v16bf load_a_frag(const bf16_t* __restrict__ p) {
    v8bf a0 = *(const v8bf*)(p);
    v8bf a1 = *(const v8bf*)(p + 16);
    return __builtin_shufflevector(a0, a1, 0,1,2,3,4,5,6,7,8,9,10,11,12,13,14,15);
}

// ---------------------------------------------------------------------------
// One RNN time step:  h_next = tanh( W_hx[x[:,t]] + b_h + h_prev @ W_hh )
// Grid: 128 blocks x 256 threads (8 waves). Wave owns 1 M-tile x 2 N-tiles.
//   blockM = blockIdx.x >> 6 (0..1), blockN = blockIdx.x & 63 (0..63)
//   mTile = blockM*8 + wave (0..15), cols [blockN*32, blockN*32+32)
// All 8 waves of a block share the same B fragments -> WGP$ broadcast reuse.
// ---------------------------------------------------------------------------
__global__ void __launch_bounds__(256)
rnn_step_kernel(const int* __restrict__ x,
                const float* __restrict__ W_hx,
                const float* __restrict__ b_h,
                const bf16_t* __restrict__ Whh_pk,
                const bf16_t* __restrict__ h_prev,
                bf16_t* __restrict__ h_next,
                int t) {
    const int lane   = threadIdx.x & 31;
    const int wave   = threadIdx.x >> 5;
    const int blockM = blockIdx.x >> 6;
    const int blockN = blockIdx.x & 63;
    const int mTile  = blockM * 8 + wave;
    const int mBase  = mTile * 16;
    const int nBase  = blockN * 32;

    const int nlo   = lane & 15;
    const int mhalf = (lane >> 4) << 3;              // 0 or 8 (C/D layout)
    const int aRow  = mBase + nlo;                   // A-operand row for this lane
    const int koff  = (lane >> 4) << 3;              // 0 or 8 (A/B operand K offset)

    // Accumulator init = gathered W_hx row + b_h (the "xw" term of this step).
    int xidx[8];
#pragma unroll
    for (int r = 0; r < 8; ++r)
        xidx[r] = x[(mBase + mhalf + r) * T_ + t];

    v8f acc[2];
#pragma unroll
    for (int i = 0; i < 2; ++i) {
        const int n = nBase + i * 16 + nlo;
        const float bh = b_h[n];
#pragma unroll
        for (int r = 0; r < 8; ++r)
            acc[i][r] = W_hx[xidx[r] * H_ + n] + bh;
    }

    // K loop: H_/32 = 64 chunks, 2 WMMA per chunk.
    const bf16_t* aBase = h_prev + aRow * H_ + koff;
    const int fragN = nBase >> 4;                    // first of 2 N fragments
#pragma unroll 4
    for (int kc = 0; kc < H_ / 32; ++kc) {
        v16bf a = load_a_frag(aBase + kc * 32);
#pragma unroll
        for (int i = 0; i < 2; ++i) {
            const bf16_t* bp = Whh_pk + (size_t)(kc * 128 + fragN + i) * 512 + lane * 16;
            v16bf b = *(const v16bf*)bp;
            acc[i] = __builtin_amdgcn_wmma_f32_16x16x32_bf16(
                false, a, false, b, (short)0, acc[i], false, false);
        }
    }

    // Epilogue: branch-free tanh -> bf16 store.
#pragma unroll
    for (int i = 0; i < 2; ++i) {
        const int n = nBase + i * 16 + nlo;
#pragma unroll
        for (int r = 0; r < 8; ++r) {
            const int b = mBase + mhalf + r;
            h_next[b * H_ + n] = (bf16_t)tanh_fast(acc[i][r]);
        }
    }
}

// ---------------------------------------------------------------------------
// Final projection: out = h_final @ W_ph + b_p   ([256,2048]@[2048,128])
// 128 tiles, 1 tile/wave -> 16 blocks x 256 threads.
// ---------------------------------------------------------------------------
__global__ void __launch_bounds__(256)
rnn_proj_kernel(const bf16_t* __restrict__ h_fin,
                const bf16_t* __restrict__ Wph_pk,
                const float* __restrict__ b_p,
                float* __restrict__ out) {
    const int lane = threadIdx.x & 31;
    const int gw   = blockIdx.x * 8 + (threadIdx.x >> 5);   // 0..127
    const int mTile = gw >> 3;                              // 0..15
    const int nt    = gw & 7;                               // 0..7
    const int mBase = mTile * 16;
    const int nBase = nt * 16;

    const int nlo   = lane & 15;
    const int mhalf = (lane >> 4) << 3;
    const int aRow  = mBase + nlo;
    const int koff  = (lane >> 4) << 3;

    v8f acc;
    const float bp = b_p[nBase + nlo];
#pragma unroll
    for (int r = 0; r < 8; ++r) acc[r] = bp;

    const bf16_t* aBase = h_fin + aRow * H_ + koff;
#pragma unroll 4
    for (int kc = 0; kc < H_ / 32; ++kc) {
        v16bf a = load_a_frag(aBase + kc * 32);
        const bf16_t* bptr = Wph_pk + (size_t)(kc * 8 + nt) * 512 + lane * 16;
        v16bf b = *(const v16bf*)bptr;
        acc = __builtin_amdgcn_wmma_f32_16x16x32_bf16(
            false, a, false, b, (short)0, acc, false, false);
    }

#pragma unroll
    for (int r = 0; r < 8; ++r) {
        const int b = mBase + mhalf + r;
        out[b * C_ + (nBase + nlo)] = acc[r];
    }
}

// ---------------------------------------------------------------------------
// Host-side orchestration (graph-capture safe: launches only).
// Workspace layout (bytes):
//   [0,              8MB)   Whh packed bf16  (H*H*2)
//   [8MB,          8.5MB)   Wph packed bf16  (H*C*2)
//   [8.5MB,        9.5MB)   h ping  bf16     (B*H*2)
//   [9.5MB,       10.5MB)   h pong  bf16     (B*H*2)
// ---------------------------------------------------------------------------
extern "C" void kernel_launch(void* const* d_in, const int* in_sizes, int n_in,
                              void* d_out, int out_size, void* d_ws, size_t ws_size,
                              hipStream_t stream) {
    const int*   x    = (const int*)d_in[0];
    const float* W_hx = (const float*)d_in[1];
    const float* W_hh = (const float*)d_in[2];
    const float* W_ph = (const float*)d_in[3];
    const float* b_h  = (const float*)d_in[4];
    const float* b_p  = (const float*)d_in[5];
    float* out = (float*)d_out;

    char* ws = (char*)d_ws;
    bf16_t* whh_pk = (bf16_t*)(ws);
    bf16_t* wph_pk = (bf16_t*)(ws + (size_t)H_ * H_ * 2);
    bf16_t* h0     = (bf16_t*)(ws + (size_t)H_ * H_ * 2 + (size_t)H_ * C_ * 2);
    bf16_t* h1     = h0 + (size_t)B_ * H_;

    // 1) Pack weights to WMMA fragment layout (bf16) and zero h0.
    pack_whh_kernel<<<(H_ * H_) / 256, 256, 0, stream>>>(W_hh, whh_pk);
    pack_wph_kernel<<<(H_ * C_) / 256, 256, 0, stream>>>(W_ph, wph_pk);
    {
        int n_dwords = (B_ * H_ * 2) / 4;
        zero_kernel<<<(n_dwords + 255) / 256, 256, 0, stream>>>((unsigned int*)h0, n_dwords);
    }

    // 2) Sequential scan: 512 step launches (inter-step sync via launch order).
    for (int t = 0; t < T_; ++t) {
        const bf16_t* hp = (t & 1) ? h1 : h0;
        bf16_t*       hn = (t & 1) ? h0 : h1;
        rnn_step_kernel<<<128, 256, 0, stream>>>(x, W_hx, b_h, whh_pk, hp, hn, t);
    }

    // 3) Projection: final h is in h0 (T even).
    rnn_proj_kernel<<<16, 256, 0, stream>>>(h0, wph_pk, b_p, out);
}
